// DiffeqSolver_20624432955592
// MI455X (gfx1250) — matrix operand order repaired and checked by reading, not compiled, not run on hardware
//
#include <hip/hip_runtime.h>
#include <cmath>

// ---------- vector types matching the gfx1250 WMMA builtin signatures ----------
typedef __attribute__((ext_vector_type(16))) __bf16 bf16x16;
typedef __attribute__((ext_vector_type(8)))  __bf16 bf16x8;
typedef __attribute__((ext_vector_type(8)))  float  f32x8;

// Problem constants (from the reference)
#define SDIM 4
#define BDIM 512
#define DDIM 256
#define HDIM 1024
#define TDIM 50
#define SB   (SDIM * BDIM)        // 2048 rows

#define LDS_STRIDE 40             // halves per LDS row = 80 bytes (16B aligned, bank-depadded)

// Branch-free tanh: exp2 + rcp (v_exp_f32 / v_rcp_f32), saturates correctly.
__device__ __forceinline__ float fast_tanh(float x) {
    float e = __builtin_amdgcn_exp2f(x * 2.8853900817779268f);  // e^{2x}
    return 1.0f - 2.0f * __builtin_amdgcn_rcpf(e + 1.0f);
}

// ---- CDNA5 async global->LDS DMA (ASYNCcnt-tracked), per-lane 16B ----
__device__ __forceinline__ void async_load_b128(unsigned lds_addr, const void* gaddr) {
    asm volatile("global_load_async_to_lds_b128 %0, %1, off"
                 :: "v"(lds_addr), "v"((unsigned long long)(uintptr_t)gaddr)
                 : "memory");
}
__device__ __forceinline__ void wait_async0() {
    asm volatile("s_wait_asynccnt 0" ::: "memory");
}

// ======================================================================
// bf16 WMMA GEMM:  C[M][N] = act(A[M][K] * Bt[N][K]^T + bias)
//   block tile 128x128, 8 wave32s, wave tile 32x64 = 2x4 WMMA 16x16 tiles
//   double-buffered LDS fed by GLOBAL_LOAD_ASYNC_TO_LDS_B128, 1 barrier/iter
// STAGE: -1 = plain bf16 activation store (hidden layers)
//        0..3 = fused RK4 stage update on the [SB][DDIM] output (f32 path)
// ======================================================================
template <bool TANH, int STAGE>
__global__ __launch_bounds__(256) void gemm_bf16_wmma(
    const __bf16* __restrict__ A,    // [M][K] row-major bf16
    const __bf16* __restrict__ Bt,   // [N][K] row-major bf16
    const float*  __restrict__ bias, // [N]
    __bf16* __restrict__ Cbf,        // [M][N] bf16 (STAGE < 0)
    float* __restrict__ accb,        // RK4 accumulator  (STAGE >= 0)
    float* __restrict__ y,           // carried state f32 (STAGE >= 0)
    __bf16* __restrict__ ytmp,       // next stage input (STAGE >= 0)
    float* __restrict__ outp,        // pred_y base      (STAGE == 3)
    const float* __restrict__ tarr, int stepIdx,
    int M, int N, int K)
{
    __shared__ __bf16 lA[2][128 * LDS_STRIDE];   // 2 x 10 KB
    __shared__ __bf16 lB[2][128 * LDS_STRIDE];   // 2 x 10 KB

    const int tid   = threadIdx.x;
    const int lane  = tid & 31;
    const int wave  = tid >> 5;
    const int bm    = blockIdx.x * 128;
    const int bn    = blockIdx.y * 128;
    const int waveM = (wave & 3) * 32;   // 4 waves tile M
    const int waveN = (wave >> 2) * 64;  // 2 waves tile N
    const int lr    = lane & 15;
    const int hi    = lane >> 4;         // lane half: 0 or 1

    // Staging geometry: 512 16B chunks per tile; thread owns chunks tid, tid+256
    // (chunk tid+256 is the same column 64 rows below).
    const int srow = tid >> 2;          // 0..63
    const int scol = (tid & 3) * 8;     // halves within row
    const __bf16* gA = A  + (size_t)(bm + srow) * K + scol;
    const __bf16* gB = Bt + (size_t)(bn + srow) * K + scol;
    const size_t rowskip = (size_t)64 * K;          // +64 rows
    const int    lds0 = srow * LDS_STRIDE + scol;
    const int    lds1 = lds0 + 64 * LDS_STRIDE;

    f32x8 acc[2][4] = {};

    // Prologue: DMA K-slice 0 into LDS buffer 0.
    async_load_b128((unsigned)(uintptr_t)&lA[0][lds0], gA);
    async_load_b128((unsigned)(uintptr_t)&lA[0][lds1], gA + rowskip);
    async_load_b128((unsigned)(uintptr_t)&lB[0][lds0], gB);
    async_load_b128((unsigned)(uintptr_t)&lB[0][lds1], gB + rowskip);

    int p = 0;
    int kk = 0;
    do {
        wait_async0();        // this wave's slice-kk DMA complete
        __syncthreads();      // everyone's DMA complete -> buffer p valid

        // ---- build WMMA fragments per ISA 16-bit layouts ----
        // A 16x32: lanes<16 hold K {0..7,16..23}; lanes>=16 hold K {8..15,24..31}
        // B 32x16: lanes<16 hold K {0..7,16..23}; lanes>=16 hold K {8..15,24..31}
        bf16x16 afrag[2], bfrag[4];
        const int khA = hi * 8;
        const int khB = hi * 8;
        #pragma unroll
        for (int mt = 0; mt < 2; ++mt) {
            const __bf16* q = &lA[p][(waveM + mt * 16 + lr) * LDS_STRIDE + khA];
            bf16x8 c0 = *(const bf16x8*)q;
            bf16x8 c1 = *(const bf16x8*)(q + 16);
            #pragma unroll
            for (int i = 0; i < 8; ++i) { afrag[mt][i] = c0[i]; afrag[mt][8 + i] = c1[i]; }
        }
        #pragma unroll
        for (int nt = 0; nt < 4; ++nt) {
            const __bf16* q = &lB[p][(waveN + nt * 16 + lr) * LDS_STRIDE + khB];
            bf16x8 c0 = *(const bf16x8*)q;
            bf16x8 c1 = *(const bf16x8*)(q + 16);
            #pragma unroll
            for (int i = 0; i < 8; ++i) { bfrag[nt][i] = c0[i]; bfrag[nt][8 + i] = c1[i]; }
        }

        // ---- DMA next K-slice into the other buffer (overlaps WMMAs) ----
        const int kn = kk + 32;
        const int np = p ^ 1;
        if (kn < K) {
            async_load_b128((unsigned)(uintptr_t)&lA[np][lds0], gA + kn);
            async_load_b128((unsigned)(uintptr_t)&lA[np][lds1], gA + rowskip + kn);
            async_load_b128((unsigned)(uintptr_t)&lB[np][lds0], gB + kn);
            async_load_b128((unsigned)(uintptr_t)&lB[np][lds1], gB + rowskip + kn);
        }

        // ---- 8 WMMAs per wave per K-step ----
        #pragma unroll
        for (int mt = 0; mt < 2; ++mt)
            #pragma unroll
            for (int nt = 0; nt < 4; ++nt)
                acc[mt][nt] = __builtin_amdgcn_wmma_f32_16x16x32_bf16(
                    false, afrag[mt], false, bfrag[nt],
                    (short)0, acc[mt][nt], false, false);

        p = np;
        kk = kn;
    } while (kk < K);

    // ---- epilogue: C/D layout (M = vgpr + 8*hi, N = lane&15) ----
    float dt = 0.0f;
    if constexpr (STAGE >= 0) dt = tarr[stepIdx + 1] - tarr[stepIdx];

    #pragma unroll
    for (int mt = 0; mt < 2; ++mt) {
        const int rbase = bm + waveM + mt * 16 + hi * 8;
        #pragma unroll
        for (int nt = 0; nt < 4; ++nt) {
            const int col = bn + waveN + nt * 16 + lr;
            const float bv = bias[col];
            #pragma unroll
            for (int r = 0; r < 8; ++r) {
                float kv = acc[mt][nt][r] + bv;
                const int row = rbase + r;
                const size_t off = (size_t)row * N + col;
                if constexpr (STAGE < 0) {
                    if constexpr (TANH) kv = fast_tanh(kv);
                    Cbf[off] = (__bf16)kv;
                } else if constexpr (STAGE == 0) {
                    accb[off] = kv;
                    ytmp[off] = (__bf16)(y[off] + 0.5f * dt * kv);
                } else if constexpr (STAGE == 1) {
                    accb[off] += 2.0f * kv;
                    ytmp[off] = (__bf16)(y[off] + 0.5f * dt * kv);
                } else if constexpr (STAGE == 2) {
                    accb[off] += 2.0f * kv;
                    ytmp[off] = (__bf16)(y[off] + dt * kv);
                } else {  // STAGE == 3
                    float a  = accb[off] + kv;
                    float yn = y[off] + (dt / 6.0f) * a;
                    y[off] = yn;
                    outp[((size_t)row * TDIM + (stepIdx + 1)) * DDIM + col] = yn;
                    ytmp[off] = (__bf16)yn;
                }
            }
        }
    }
}

// ======================================================================
// Weight prep: Wt[n][k] = bf16(W[k][n])
// ======================================================================
__global__ __launch_bounds__(256) void transpose_to_bf16(
    const float* __restrict__ W, __bf16* __restrict__ Wt, int K, int N)
{
    int idx = blockIdx.x * 256 + threadIdx.x;
    if (idx >= K * N) return;
    int k = idx / N, n = idx - k * N;          // coalesced read along N
    Wt[(size_t)n * K + k] = (__bf16)W[idx];
}

// ======================================================================
// Init: y = first_point (f32), ytmp = bf16(y), pred_y[:, :, 0, :] = y,
//       reg_state (last SDIM floats of d_out) = 0
// ======================================================================
__global__ __launch_bounds__(256) void init_state(
    const float* __restrict__ first, float* __restrict__ y,
    __bf16* __restrict__ ytmp, float* __restrict__ out, size_t predElems)
{
    int i = blockIdx.x * 256 + threadIdx.x;    // grid == SB*DDIM exactly
    float v = first[i];
    y[i] = v;
    ytmp[i] = (__bf16)v;
    int r = i / DDIM, d = i - r * DDIM;
    out[(size_t)r * TDIM * DDIM + d] = v;
    if (i < SDIM) out[predElems + i] = 0.0f;
}

// ======================================================================
// Host-side orchestration (all launches on `stream`; graph-capture safe)
// ======================================================================
extern "C" void kernel_launch(void* const* d_in, const int* in_sizes, int n_in,
                              void* d_out, int out_size, void* d_ws, size_t ws_size,
                              hipStream_t stream)
{
    const float* first = (const float*)d_in[0];
    const float* tarr  = (const float*)d_in[1];
    const float* W1 = (const float*)d_in[2]; const float* b1 = (const float*)d_in[3];
    const float* W2 = (const float*)d_in[4]; const float* b2 = (const float*)d_in[5];
    const float* W3 = (const float*)d_in[6]; const float* b3 = (const float*)d_in[7];
    float* out = (float*)d_out;

    // Workspace layout (offsets are 256B-aligned; total ~16.8 MB)
    char* ws = (char*)d_ws;
    __bf16* W1t  = (__bf16*)(ws + 0);                    // [H][D]  bf16  512 KB
    __bf16* W2t  = (__bf16*)(ws + 524288);               // [H][H]  bf16    2 MB
    __bf16* W3t  = (__bf16*)(ws + 2621440);              // [D][H]  bf16  512 KB
    float*  y    = (float*) (ws + 3145728);              // [SB][D] f32     2 MB
    __bf16* ytmp = (__bf16*)(ws + 5242880);              // [SB][D] bf16    1 MB
    __bf16* h1   = (__bf16*)(ws + 6291456);              // [SB][H] bf16    4 MB
    __bf16* h2   = (__bf16*)(ws + 10485760);             // [SB][H] bf16    4 MB
    float*  accb = (float*) (ws + 14680064);             // [SB][D] f32     2 MB

    const size_t predElems = (size_t)SB * TDIM * DDIM;   // pred_y element count
    const int nElem = SB * DDIM;                         // 524288, % 256 == 0

    // One-time (per launch) weight transpose + downconvert: stays hot in 192MB L2.
    transpose_to_bf16<<<(DDIM * HDIM + 255) / 256, 256, 0, stream>>>(W1, W1t, DDIM, HDIM);
    transpose_to_bf16<<<(HDIM * HDIM + 255) / 256, 256, 0, stream>>>(W2, W2t, HDIM, HDIM);
    transpose_to_bf16<<<(HDIM * DDIM + 255) / 256, 256, 0, stream>>>(W3, W3t, HDIM, DDIM);

    init_state<<<nElem / 256, 256, 0, stream>>>(first, y, ytmp, out, predElems);

    const dim3 gBig (SB / 128, HDIM / 128);  // 16 x 8
    const dim3 gSmall(SB / 128, DDIM / 128); // 16 x 2

    for (int step = 0; step < TDIM - 1; ++step) {
        #define RK4_STAGE(ST)                                                                 \
            gemm_bf16_wmma<true, -1><<<gBig, 256, 0, stream>>>(                               \
                ytmp, W1t, b1, h1, nullptr, nullptr, nullptr, nullptr, nullptr, 0,            \
                SB, HDIM, DDIM);                                                              \
            gemm_bf16_wmma<true, -1><<<gBig, 256, 0, stream>>>(                               \
                h1, W2t, b2, h2, nullptr, nullptr, nullptr, nullptr, nullptr, 0,              \
                SB, HDIM, HDIM);                                                              \
            gemm_bf16_wmma<false, ST><<<gSmall, 256, 0, stream>>>(                            \
                h2, W3t, b3, nullptr, accb, y, ytmp, out, tarr, step,                         \
                SB, DDIM, HDIM);
        RK4_STAGE(0)
        RK4_STAGE(1)
        RK4_STAGE(2)
        RK4_STAGE(3)
        #undef RK4_STAGE
    }
}
